// BitBertMLP_46377056862386
// MI455X (gfx1250) — 8-bit matrix operand order repaired and checked by reading, compiled, not run on hardware
//
#include <hip/hip_runtime.h>
#include <hip/hip_bf16.h>
#include <stdint.h>


typedef __attribute__((ext_vector_type(8))) int v8i;

#define NTOK 32768
#define DIM  768
#define HID  2048
#define O1   (2 * HID)

#define OFF_SUMS   0u
#define OFF_QWIN   256u
#define OFF_QWOUT  (OFF_QWIN  + 3145728u)
#define OFF_QX     (OFF_QWOUT + 1572864u)
#define OFF_SX     (OFF_QX    + 25165824u)
#define OFF_QY     (OFF_SX    + 131072u)
#define OFF_SY     (OFF_QY    + 67108864u)
#define OFF_Y      (OFF_SY    + 131072u)

__global__ __launch_bounds__(256) void k_abssum(const float* __restrict__ w,
                                                int n, float* __restrict__ sum) {
    __shared__ float red[256];
    float acc = 0.f;
    for (int i = blockIdx.x * 256 + threadIdx.x; i < n; i += gridDim.x * 256)
        acc += fabsf(w[i]);
    red[threadIdx.x] = acc;
    __syncthreads();
    for (int s = 128; s > 0; s >>= 1) {
        if ((int)threadIdx.x < s) red[threadIdx.x] += red[threadIdx.x + s];
        __syncthreads();
    }
    if (threadIdx.x == 0) atomicAdd(sum, red[0]);
}

__global__ __launch_bounds__(256) void k_quant_w(const float* __restrict__ w, int n,
                                                 const float* __restrict__ sum,
                                                 float invn,
                                                 int8_t* __restrict__ q) {
    float mean = *sum * invn;
    float s = 1.f / fmaxf(mean, 1e-5f);
    for (int i = blockIdx.x * 256 + threadIdx.x; i < n; i += gridDim.x * 256) {
        float v = rintf(w[i] * s);
        v = fminf(fmaxf(v, -1.f), 1.f);
        q[i] = (int8_t)v;
    }
}

__device__ __forceinline__ float ld_as_f32(const float* p)          { return *p; }
__device__ __forceinline__ float ld_as_f32(const __hip_bfloat16* p) { return __bfloat162float(*p); }

template <int L, typename T>
__global__ __launch_bounds__(256) void k_actquant(const T* __restrict__ x,
                                                  const float* __restrict__ g,
                                                  int8_t* __restrict__ q,
                                                  float* __restrict__ sx) {
    constexpr int EPT = L / 256;
    const int tok = blockIdx.x;
    const int tid = threadIdx.x;
    const T* row = x + (size_t)tok * L;

    __shared__ float red[256];
    __shared__ float bc;

    float vals[EPT];
    float ss = 0.f;
#pragma unroll
    for (int i = 0; i < EPT; ++i) {
        float v = ld_as_f32(row + tid + 256 * i);
        vals[i] = v;
        ss += v * v;
    }
    red[tid] = ss;
    __syncthreads();
    for (int s = 128; s > 0; s >>= 1) {
        if (tid < s) red[tid] += red[tid + s];
        __syncthreads();
    }
    if (tid == 0) bc = rsqrtf(red[0] * (1.f / (float)L) + 1e-6f);
    __syncthreads();
    const float rms = bc;
    __syncthreads();

    float amax = 0.f;
#pragma unroll
    for (int i = 0; i < EPT; ++i) {
        float xn = vals[i] * rms * g[tid + 256 * i];
        vals[i] = xn;
        amax = fmaxf(amax, fabsf(xn));
    }
    red[tid] = amax;
    __syncthreads();
    for (int s = 128; s > 0; s >>= 1) {
        if (tid < s) red[tid] = fmaxf(red[tid], red[tid + s]);
        __syncthreads();
    }
    if (tid == 0) bc = 127.f / fmaxf(red[0], 1e-5f);
    __syncthreads();
    const float sc = bc;

#pragma unroll
    for (int i = 0; i < EPT; ++i) {
        float qv = fminf(fmaxf(rintf(vals[i] * sc), -128.f), 127.f);
        q[(size_t)tok * L + tid + 256 * i] = (int8_t)qv;
    }
    if (tid == 0) sx[tok] = sc;
}

__device__ __forceinline__ v8i load_fragA(const int8_t* p  ) {
    int4 p0 = *(const int4*)(p);
    int4 p1 = *(const int4*)(p + 32);
    v8i a;
    a[0] = p0.x; a[1] = p0.y; a[2] = p0.z; a[3] = p0.w;
    a[4] = p1.x; a[5] = p1.y; a[6] = p1.z; a[7] = p1.w;
    return a;
}
__device__ __forceinline__ v8i load_fragB(const int8_t* p  ) {
    int4 b0 = *(const int4*)(p);
    int4 b1 = *(const int4*)(p + 32);
    v8i b;
    b[0] = b0.x; b[1] = b0.y; b[2] = b0.z; b[3] = b0.w;
    b[4] = b1.x; b[5] = b1.y; b[6] = b1.z; b[7] = b1.w;
    return b;
}
#define WMMA_IU8(A, B, C) \
    __builtin_amdgcn_wmma_i32_16x16x64_iu8(true, (A), true, (B), (C), false, false)

__global__ __launch_bounds__(256) void k_gemm1_swiglu(const int8_t* __restrict__ qx,
                                                      const float*  __restrict__ sx,
                                                      const int8_t* __restrict__ qw,
                                                      const float*  __restrict__ wsum,
                                                      __hip_bfloat16* __restrict__ y) {
    const int lane = threadIdx.x & 31;
    const int wid  = threadIdx.x >> 5;
    const int tile = blockIdx.x * 8 + wid;
    const int n2 = tile & 63;
    const int m2 = tile >> 6;
    const int hi   = (lane & 16) ? 1 : 0;
    const int l15  = lane & 15;

    const int row0 = m2 * 32 + l15;
    const int cu0  = n2 * 32 + l15;

    const int8_t* a0 = qx + (size_t)row0 * DIM + (hi ? 16 : 0);
    const int8_t* a1 = a0 + (size_t)16 * DIM;
    const int8_t* bu0 = qw + (size_t)cu0 * DIM + (hi ? 16 : 0);
    const int8_t* bu1 = bu0 + (size_t)16 * DIM;
    const int8_t* bg0 = qw + (size_t)(cu0 + HID) * DIM + (hi ? 16 : 0);
    const int8_t* bg1 = bg0 + (size_t)16 * DIM;

    v8i u00 = {0,0,0,0,0,0,0,0}, u01 = u00, u10 = u00, u11 = u00;
    v8i g00 = u00, g01 = u00, g10 = u00, g11 = u00;

#pragma unroll 2
    for (int k0 = 0; k0 < DIM; k0 += 64) {
        v8i A0 = load_fragA(a0 + k0);
        v8i A1 = load_fragA(a1 + k0);
        v8i BU0 = load_fragB(bu0 + k0);
        v8i BU1 = load_fragB(bu1 + k0);
        v8i BG0 = load_fragB(bg0 + k0);
        v8i BG1 = load_fragB(bg1 + k0);
        u00 = WMMA_IU8(A0, BU0, u00);
        u01 = WMMA_IU8(A0, BU1, u01);
        u10 = WMMA_IU8(A1, BU0, u10);
        u11 = WMMA_IU8(A1, BU1, u11);
        g00 = WMMA_IU8(A0, BG0, g00);
        g01 = WMMA_IU8(A0, BG1, g01);
        g10 = WMMA_IU8(A1, BG0, g10);
        g11 = WMMA_IU8(A1, BG1, g11);
    }

    const float fac = fmaxf(wsum[0] * (1.f / (float)(O1 * DIM)), 1e-5f);
    const int rbase0 = m2 * 32 + (hi ? 8 : 0);
    const int rbase1 = rbase0 + 16;
    const int c0 = n2 * 32 + l15;
    const int c1 = c0 + 16;
#pragma unroll
    for (int r = 0; r < 8; ++r) {
        {
            int row = rbase0 + r;
            float inv = fac / sx[row];
            float u_0 = (float)u00[r] * inv, g_0 = (float)g00[r] * inv;
            float u_1 = (float)u01[r] * inv, g_1 = (float)g01[r] * inv;
            float s0 = g_0 / (1.f + __expf(-g_0));
            float s1 = g_1 / (1.f + __expf(-g_1));
            y[(size_t)row * HID + c0] = __float2bfloat16(s0 * u_0);
            y[(size_t)row * HID + c1] = __float2bfloat16(s1 * u_1);
        }
        {
            int row = rbase1 + r;
            float inv = fac / sx[row];
            float u_0 = (float)u10[r] * inv, g_0 = (float)g10[r] * inv;
            float u_1 = (float)u11[r] * inv, g_1 = (float)g11[r] * inv;
            float s0 = g_0 / (1.f + __expf(-g_0));
            float s1 = g_1 / (1.f + __expf(-g_1));
            y[(size_t)row * HID + c0] = __float2bfloat16(s0 * u_0);
            y[(size_t)row * HID + c1] = __float2bfloat16(s1 * u_1);
        }
    }
}

__global__ __launch_bounds__(256) void k_gemm2(const int8_t* __restrict__ qy,
                                               const float*  __restrict__ sy,
                                               const int8_t* __restrict__ qw,
                                               const float*  __restrict__ wsum,
                                               float* __restrict__ out) {
    const int lane = threadIdx.x & 31;
    const int wid  = threadIdx.x >> 5;
    const int tile = blockIdx.x * 8 + wid;
    const int n2 = tile % 24;
    const int m2 = tile / 24;
    const int hi   = (lane & 16) ? 1 : 0;
    const int l15  = lane & 15;

    const int row0 = m2 * 32 + l15;
    const int col0 = n2 * 32 + l15;

    const int8_t* a0 = qy + (size_t)row0 * HID + (hi ? 16 : 0);
    const int8_t* a1 = a0 + (size_t)16 * HID;
    const int8_t* b0p = qw + (size_t)col0 * HID + (hi ? 16 : 0);
    const int8_t* b1p = b0p + (size_t)16 * HID;

    v8i c00 = {0,0,0,0,0,0,0,0}, c01 = c00, c10 = c00, c11 = c00;

#pragma unroll 2
    for (int k0 = 0; k0 < HID; k0 += 64) {
        v8i A0 = load_fragA(a0 + k0);
        v8i A1 = load_fragA(a1 + k0);
        v8i B0 = load_fragB(b0p + k0);
        v8i B1 = load_fragB(b1p + k0);
        c00 = WMMA_IU8(A0, B0, c00);
        c01 = WMMA_IU8(A0, B1, c01);
        c10 = WMMA_IU8(A1, B0, c10);
        c11 = WMMA_IU8(A1, B1, c11);
    }

    const float fac = fmaxf(wsum[1] * (1.f / (float)(DIM * HID)), 1e-5f);
    const int rbase0 = m2 * 32 + (hi ? 8 : 0);
    const int rbase1 = rbase0 + 16;
    const int cc0 = n2 * 32 + l15;
    const int cc1 = cc0 + 16;
#pragma unroll
    for (int r = 0; r < 8; ++r) {
        {
            int row = rbase0 + r;
            float inv = fac / sy[row];
            out[(size_t)row * DIM + cc0] = (float)c00[r] * inv;
            out[(size_t)row * DIM + cc1] = (float)c01[r] * inv;
        }
        {
            int row = rbase1 + r;
            float inv = fac / sy[row];
            out[(size_t)row * DIM + cc0] = (float)c10[r] * inv;
            out[(size_t)row * DIM + cc1] = (float)c11[r] * inv;
        }
    }
}

extern "C" void kernel_launch(void* const* d_in, const int* in_sizes, int n_in,
                              void* d_out, int out_size, void* d_ws, size_t ws_size,
                              hipStream_t stream) {
    (void)in_sizes; (void)n_in; (void)out_size; (void)ws_size;

    const float* x     = (const float*)d_in[0];
    const float* w_in  = (const float*)d_in[1];
    const float* g_in  = (const float*)d_in[2];
    const float* w_out = (const float*)d_in[3];
    const float* g_out = (const float*)d_in[4];
    float* out = (float*)d_out;

    uint8_t* ws = (uint8_t*)d_ws;
    float*   sums   = (float*)(ws + OFF_SUMS);
    int8_t*  qw_in  = (int8_t*)(ws + OFF_QWIN);
    int8_t*  qw_out = (int8_t*)(ws + OFF_QWOUT);
    int8_t*  qx     = (int8_t*)(ws + OFF_QX);
    float*   sx     = (float*)(ws + OFF_SX);
    int8_t*  qy     = (int8_t*)(ws + OFF_QY);
    float*   sy     = (float*)(ws + OFF_SY);
    __hip_bfloat16* y = (__hip_bfloat16*)(ws + OFF_Y);

    const int n_win  = O1 * DIM;
    const int n_wout = DIM * HID;

    hipMemsetAsync(sums, 0, 2 * sizeof(float), stream);

    k_abssum<<<1024, 256, 0, stream>>>(w_in,  n_win,  sums + 0);
    k_abssum<<<1024, 256, 0, stream>>>(w_out, n_wout, sums + 1);

    k_quant_w<<<3072, 256, 0, stream>>>(w_in,  n_win,  sums + 0, 1.f / (float)n_win,  qw_in);
    k_quant_w<<<1536, 256, 0, stream>>>(w_out, n_wout, sums + 1, 1.f / (float)n_wout, qw_out);

    k_actquant<DIM, float><<<NTOK, 256, 0, stream>>>(x, g_in, qx, sx);

    k_gemm1_swiglu<<<8192, 256, 0, stream>>>(qx, sx, qw_in, sums, y);

    k_actquant<HID, __hip_bfloat16><<<NTOK, 256, 0, stream>>>(y, g_out, qy, sy);

    k_gemm2<<<3072, 256, 0, stream>>>(qy, sy, qw_out, sums, out);
}
